// matchGAT3_2353642078848
// MI455X (gfx1250) — matrix operand order repaired and checked by reading, not compiled, not run on hardware
//
#include <hip/hip_runtime.h>
#include <stdint.h>

#define NNODES 10000
#define NIN    256
#define NOUT   128
#define NEDGE  320000

typedef __attribute__((ext_vector_type(16))) __bf16 v16bf;
typedef __attribute__((ext_vector_type(8)))  float  v8f;

// ---------- helpers ----------
__device__ __forceinline__ unsigned short f2bfbits(float f) {
  union { float f; unsigned u; } v; v.f = f;
  unsigned r = v.u + 0x7FFFu + ((v.u >> 16) & 1u);     // RNE
  return (unsigned short)(r >> 16);
}
// order-preserving float <-> uint encoding for atomicMax on floats
__device__ __forceinline__ unsigned f2ord(float f) {
  unsigned u = __float_as_uint(f);
  return (u & 0x80000000u) ? ~u : (u | 0x80000000u);
}
__device__ __forceinline__ float ord2f(unsigned u) {
  u = (u & 0x80000000u) ? (u & 0x7FFFFFFFu) : ~u;
  return __uint_as_float(u);
}

// ---------- bulk f32 -> bf16 conversion (done ONCE per activation matrix) ----------
__global__ void f32_to_bf16(const float* __restrict__ in,
                            unsigned short* __restrict__ out, int n8)
{
  const int i = blockIdx.x * blockDim.x + threadIdx.x;
  if (i >= n8) return;
  const float4 f0 = ((const float4*)in)[2 * i];
  const float4 f1 = ((const float4*)in)[2 * i + 1];
  union { unsigned short h[8]; uint4 u; } p;
  p.h[0] = f2bfbits(f0.x); p.h[1] = f2bfbits(f0.y);
  p.h[2] = f2bfbits(f0.z); p.h[3] = f2bfbits(f0.w);
  p.h[4] = f2bfbits(f1.x); p.h[5] = f2bfbits(f1.y);
  p.h[6] = f2bfbits(f1.z); p.h[7] = f2bfbits(f1.w);
  ((uint4*)out)[i] = p.u;
}

// ---------- pack weights into WMMA B-fragment order ----------
// lane = half*16 + l15 ; lane's 16 bf16 = W[(kt*32 + half*16 + e)*NOUT + nt*16 + l15]
// packed flat: (((kt*8 + nt)*32 + lane)*16 + e)
__global__ void pack_w_bf16(int K,
    const float* __restrict__ W0, unsigned short* __restrict__ P0,
    const float* __restrict__ W1, unsigned short* __restrict__ P1,
    const float* __restrict__ W2, unsigned short* __restrict__ P2,
    const float* __restrict__ W3, unsigned short* __restrict__ P3)
{
  const int z = blockIdx.y;
  const float*    W = (z == 0) ? W0 : (z == 1) ? W1 : (z == 2) ? W2 : W3;
  unsigned short* P = (z == 0) ? P0 : (z == 1) ? P1 : (z == 2) ? P2 : P3;

  const int t = blockIdx.x * blockDim.x + threadIdx.x;   // (kt*8 + nt)*32 + lane
  if (t >= (K >> 5) * 256) return;
  const int lane = t & 31, nt = (t >> 5) & 7, kt = t >> 8;
  const int half = lane >> 4, l15 = lane & 15;
  const int col   = nt * 16 + l15;
  const int kbase = kt * 32 + half * 16;

  union { unsigned short h[16]; uint4 u[2]; } p;
#pragma unroll
  for (int e = 0; e < 16; ++e)
    p.h[e] = f2bfbits(W[(size_t)(kbase + e) * NOUT + col]);
  uint4* dst = (uint4*)(P + (size_t)t * 16);
  dst[0] = p.u[0]; dst[1] = p.u[1];
}

// ---------- 4-way projection GEMM: Y_z = Xbf @ W_z + B_z ----------
// one wave per 16-row strip; A fragment loaded once per k-step, reused across
// 8 N-tiles -> 8 back-to-back v_wmma_f32_16x16x32_bf16. All fragment loads are
// issued off a single base address with immediate offsets (B: nt*1024 {+16},
// kt step folded into base), so the hot loop is loads + wmma only.
__global__ void gemm4_wmma(const unsigned short* __restrict__ Xbf, int K,
    const unsigned short* __restrict__ P0, const float* __restrict__ B0, float* __restrict__ Y0,
    const unsigned short* __restrict__ P1, const float* __restrict__ B1, float* __restrict__ Y1,
    const unsigned short* __restrict__ P2, const float* __restrict__ B2, float* __restrict__ Y2,
    const unsigned short* __restrict__ P3, const float* __restrict__ B3, float* __restrict__ Y3)
{
  const int z = blockIdx.y;
  const unsigned short* P  = (z == 0) ? P0 : (z == 1) ? P1 : (z == 2) ? P2 : P3;
  const float*          Bb = (z == 0) ? B0 : (z == 1) ? B1 : (z == 2) ? B2 : B3;
  float*                Y  = (z == 0) ? Y0 : (z == 1) ? Y1 : (z == 2) ? Y2 : Y3;

  const int lane = threadIdx.x & 31;
  const int half = lane >> 4, l15 = lane & 15;
  const int mrow = blockIdx.x * 16 + l15;

  v8f zero = {};
  v8f acc[8];
#pragma unroll
  for (int nt = 0; nt < 8; ++nt) acc[nt] = zero;

  // per-lane bases; everything inside the loop is base + constant
  const char* abase = (const char*)Xbf + ((size_t)mrow * K + half * 16) * 2;
  const char* bbase = (const char*)P + (size_t)lane * 32;

  const int ksteps = K >> 5;
  for (int kt = 0; kt < ksteps; ++kt) {
    // A 16x32 bf16 fragment: elems 0..15 -> K=kbase+half*16..+15 (contiguous, matches packed B)
    union { uint4 u[2]; v16bf v; } fa;
    const char* ar = abase + (size_t)kt * 64;
    fa.u[0] = *(const uint4*)(ar);
    fa.u[1] = *(const uint4*)(ar + 16);
    const char* bk = bbase + (size_t)kt * 8192;
#pragma unroll
    for (int nt = 0; nt < 8; ++nt) {
      union { uint4 u[2]; v16bf v; } fb;
      fb.u[0] = *(const uint4*)(bk + nt * 1024);
      fb.u[1] = *(const uint4*)(bk + nt * 1024 + 16);
      acc[nt] = __builtin_amdgcn_wmma_f32_16x16x32_bf16(
          false, fa.v, false, fb.v, (short)0, acc[nt], false, false);
    }
  }
#pragma unroll
  for (int nt = 0; nt < 8; ++nt) {
    const int col = nt * 16 + l15;
    const float bias = Bb[col];
#pragma unroll
    for (int r = 0; r < 8; ++r)
      Y[(size_t)(blockIdx.x * 16 + r + half * 8) * NOUT + col] = acc[nt][r] + bias;
  }
}

// ---------- attention plumbing ----------
__global__ void init_attn(const float* __restrict__ S, float* __restrict__ H,
                          unsigned* __restrict__ mbits, float* __restrict__ denom)
{
  const int i = blockIdx.x * blockDim.x + threadIdx.x;
  if (i < NNODES * NOUT) H[i] = S[i];            // seed aggregation with skip branch
  if (i < NNODES) { mbits[i] = 0x007FFFFFu; denom[i] = 0.0f; }  // f2ord(-inf)
}

__global__ void edge_logits(const float* __restrict__ Q, const float* __restrict__ Kf,
                            const long long* __restrict__ ei,
                            float* __restrict__ logits, unsigned* __restrict__ mbits)
{
  const int w = blockIdx.x * (blockDim.x >> 5) + (threadIdx.x >> 5);
  const int lane = threadIdx.x & 31;
  if (w >= NEDGE) return;
  const long long s = ei[w];
  const long long t = ei[NEDGE + w];
  const float* q = Q  + (size_t)t * NOUT;
  const float* k = Kf + (size_t)s * NOUT;
  float p = 0.f;
#pragma unroll
  for (int f = 0; f < 4; ++f) p += q[lane + 32 * f] * k[lane + 32 * f];
#pragma unroll
  for (int off = 16; off; off >>= 1) p += __shfl_xor(p, off, 32);
  if (lane == 0) {
    p *= 0.08838834764831845f;   // 1/sqrt(128)
    logits[w] = p;
    atomicMax(&mbits[t], f2ord(p));
  }
}

__global__ void edge_expsum(float* __restrict__ logits, const long long* __restrict__ ei,
                            const unsigned* __restrict__ mbits, float* __restrict__ denom)
{
  const int e = blockIdx.x * blockDim.x + threadIdx.x;
  if (e >= NEDGE) return;
  const long long t = ei[NEDGE + e];
  const float v = __expf(logits[e] - ord2f(mbits[t]));
  logits[e] = v;
  atomicAdd(&denom[t], v);
}

__global__ void edge_aggregate(const float* __restrict__ V, const long long* __restrict__ ei,
                               const float* __restrict__ logits, const float* __restrict__ denom,
                               float* __restrict__ H)
{
  const int w = blockIdx.x * (blockDim.x >> 5) + (threadIdx.x >> 5);
  const int lane = threadIdx.x & 31;
  if (w >= NEDGE) return;
  const long long s = ei[w];
  const long long t = ei[NEDGE + w];
  const float a = logits[w] / denom[t];
  const float* vs = V + (size_t)s * NOUT;
  float*       ht = H + (size_t)t * NOUT;
#pragma unroll
  for (int f = 0; f < 4; ++f)
    atomicAdd(&ht[lane + 32 * f], a * vs[lane + 32 * f]);
}

__global__ void relu_ip(float* __restrict__ H, int n) {
  const int i = blockIdx.x * blockDim.x + threadIdx.x;
  if (i < n) H[i] = fmaxf(H[i], 0.f);
}

// ---------- collapse MLP heads: wl = l1w@l2w, bl = l1b@l2w + l2b (same for r) ----------
__global__ void head_collapse(const float* __restrict__ l1w, const float* __restrict__ l1b,
                              const float* __restrict__ l2w, const float* __restrict__ l2b,
                              const float* __restrict__ r1w, const float* __restrict__ r1b,
                              const float* __restrict__ r2w, const float* __restrict__ r2b,
                              float* __restrict__ wl, float* __restrict__ wr,
                              float* __restrict__ beff)
{
  const int tid = threadIdx.x;
  if (tid < 128) {
    float s = 0.f;
    for (int j = 0; j < 128; ++j) s += l1w[tid * 128 + j] * l2w[j];
    wl[tid] = s;
    if (tid == 0) { float b = 0.f; for (int j = 0; j < 128; ++j) b += l1b[j] * l2w[j]; beff[0] = b + l2b[0]; }
  } else {
    const int k = tid - 128;
    float s = 0.f;
    for (int j = 0; j < 128; ++j) s += r1w[k * 128 + j] * r2w[j];
    wr[k] = s;
    if (k == 0) { float b = 0.f; for (int j = 0; j < 128; ++j) b += r1b[j] * r2w[j]; beff[1] = b + r2b[0]; }
  }
}

// ---------- c[i] = 0.5*((F[i]@wl+bl) + (F[i]@wr+br)) ----------
__global__ void node_scores(const float* __restrict__ F, const float* __restrict__ wl,
                            const float* __restrict__ wr, const float* __restrict__ beff,
                            float* __restrict__ c)
{
  const int node = blockIdx.x * (blockDim.x >> 5) + (threadIdx.x >> 5);
  const int lane = threadIdx.x & 31;
  if (node >= NNODES) return;
  const float* f = F + (size_t)node * NOUT;
  float pl = 0.f, pr = 0.f;
#pragma unroll
  for (int i = 0; i < 4; ++i) {
    const float x = f[lane + 32 * i];
    pl += x * wl[lane + 32 * i];
    pr += x * wr[lane + 32 * i];
  }
#pragma unroll
  for (int off = 16; off; off >>= 1) { pl += __shfl_xor(pl, off, 32); pr += __shfl_xor(pr, off, 32); }
  if (lane == 0) c[node] = 0.5f * ((pl + beff[0]) + (pr + beff[1]));
}

// ---------- out[i,j] = sigmoid(c[i] + c[j]) : 400 MB, float4 stores ----------
__global__ void pairwise_sigmoid(const float* __restrict__ c, float* __restrict__ out)
{
  const int i  = blockIdx.y;
  const int j0 = (blockIdx.x * blockDim.x + threadIdx.x) * 4;   // NNODES % 4 == 0
  if (j0 >= NNODES) return;
  const float ci = c[i];
  const float4 cv = *(const float4*)(c + j0);
  float4 r;
  r.x = 1.f / (1.f + __expf(-(ci + cv.x)));
  r.y = 1.f / (1.f + __expf(-(ci + cv.y)));
  r.z = 1.f / (1.f + __expf(-(ci + cv.z)));
  r.w = 1.f / (1.f + __expf(-(ci + cv.w)));
  *(float4*)(out + (size_t)i * NNODES + j0) = r;
}

// ---------- launcher ----------
extern "C" void kernel_launch(void* const* d_in, const int* in_sizes, int n_in,
                              void* d_out, int out_size, void* d_ws, size_t ws_size,
                              hipStream_t stream) {
  const float*     x   = (const float*)d_in[0];
  const long long* ei  = (const long long*)d_in[1];   // int64 edge_index [2,E]
  const float *k1w = (const float*)d_in[2],  *k1b = (const float*)d_in[3];
  const float *q1w = (const float*)d_in[4],  *q1b = (const float*)d_in[5];
  const float *v1w = (const float*)d_in[6],  *v1b = (const float*)d_in[7];
  const float *s1w = (const float*)d_in[8],  *s1b = (const float*)d_in[9];
  const float *k2w = (const float*)d_in[10], *k2b = (const float*)d_in[11];
  const float *q2w = (const float*)d_in[12], *q2b = (const float*)d_in[13];
  const float *v2w = (const float*)d_in[14], *v2b = (const float*)d_in[15];
  const float *s2w = (const float*)d_in[16], *s2b = (const float*)d_in[17];
  const float *l1w = (const float*)d_in[18], *l1b = (const float*)d_in[19];
  const float *l2w = (const float*)d_in[20], *l2b = (const float*)d_in[21];
  const float *r1w = (const float*)d_in[22], *r1b = (const float*)d_in[23];
  const float *r2w = (const float*)d_in[24], *r2b = (const float*)d_in[25];

  float* ws = (float*)d_ws;
  const size_t SZ = (size_t)NNODES * NOUT;             // 1,280,000 floats
  float*    Q      = ws;
  float*    K      = ws + 1 * SZ;
  float*    V      = ws + 2 * SZ;
  float*    S      = ws + 3 * SZ;
  float*    H      = ws + 4 * SZ;                      // layer-1 aggregation -> h
  float*    F      = ws + 5 * SZ;                      // layer-2 aggregation -> feats
  float*    logits = ws + 6 * SZ;                      // [E]
  float*    denom  = logits + NEDGE;                   // [N]
  unsigned* mbits  = (unsigned*)(denom + 10240);       // [N]
  float*    c      = (float*)mbits + 10240;            // [N], 16B-aligned
  float*    wl     = c + 10240;
  float*    wr     = wl + 128;
  float*    beff   = wr + 128;
  // bf16 staging regions (float-offset based, 16B aligned)
  unsigned short* Xbf = (unsigned short*)(ws + 8031232);            // up to 2.56M bf16
  unsigned short* P0  = (unsigned short*)(ws + 8031232 + 1280000);  // 4 x K*128 bf16
  unsigned short* P1  = P0 + (size_t)NIN * NOUT;
  unsigned short* P2  = P1 + (size_t)NIN * NOUT;
  unsigned short* P3  = P2 + (size_t)NIN * NOUT;

  const dim3 gemmGrid(NNODES / 16, 4);                 // 625 m-strips x 4 weights
  const int  nfBlocks = (NNODES * NOUT + 255) / 256;   // 5000
  const int  waveBlocks = NEDGE / 8;                   // 40000 (8 waves / 256-thread block)

  // ----- layer 1 TransformerConv -----
  f32_to_bf16<<<(NNODES * NIN / 8 + 255) / 256, 256, 0, stream>>>(x, Xbf, NNODES * NIN / 8);
  pack_w_bf16<<<dim3(8, 4), 256, 0, stream>>>(NIN, q1w, P0, k1w, P1, v1w, P2, s1w, P3);
  gemm4_wmma<<<gemmGrid, 32, 0, stream>>>(Xbf, NIN,
      P0, q1b, Q,  P1, k1b, K,  P2, v1b, V,  P3, s1b, S);
  init_attn<<<nfBlocks, 256, 0, stream>>>(S, H, mbits, denom);
  edge_logits<<<waveBlocks, 256, 0, stream>>>(Q, K, ei, logits, mbits);
  edge_expsum<<<(NEDGE + 255) / 256, 256, 0, stream>>>(logits, ei, mbits, denom);
  edge_aggregate<<<waveBlocks, 256, 0, stream>>>(V, ei, logits, denom, H);
  relu_ip<<<nfBlocks, 256, 0, stream>>>(H, NNODES * NOUT);

  // ----- layer 2 TransformerConv (reuse Q/K/V/S, Xbf, P0..P3) -----
  f32_to_bf16<<<(NNODES * NOUT / 8 + 255) / 256, 256, 0, stream>>>(H, Xbf, NNODES * NOUT / 8);
  pack_w_bf16<<<dim3(4, 4), 256, 0, stream>>>(NOUT, q2w, P0, k2w, P1, v2w, P2, s2w, P3);
  gemm4_wmma<<<gemmGrid, 32, 0, stream>>>(Xbf, NOUT,
      P0, q2b, Q,  P1, k2b, K,  P2, v2b, V,  P3, s2b, S);
  init_attn<<<nfBlocks, 256, 0, stream>>>(S, F, mbits, denom);
  edge_logits<<<waveBlocks, 256, 0, stream>>>(Q, K, ei, logits, mbits);
  edge_expsum<<<(NEDGE + 255) / 256, 256, 0, stream>>>(logits, ei, mbits, denom);
  edge_aggregate<<<waveBlocks, 256, 0, stream>>>(V, ei, logits, denom, F);

  // ----- rank-1 head + 400MB sigmoid write -----
  head_collapse<<<1, 256, 0, stream>>>(l1w, l1b, l2w, l2b, r1w, r1b, r2w, r2b, wl, wr, beff);
  node_scores<<<(NNODES + 7) / 8, 256, 0, stream>>>(F, wl, wr, beff, c);
  pairwise_sigmoid<<<dim3(10, NNODES), 256, 0, stream>>>(c, (float*)d_out);
}
